// Block_50706383897045
// MI455X (gfx1250) — matrix operand order repaired and checked by reading, not compiled, not run on hardware
//
#include <hip/hip_runtime.h>
#include <hip/hip_bf16.h>
#include <math.h>
#include <stdint.h>

// ---------------- problem dims ----------------
#define BQ   8
#define TT   1024
#define CC   768
#define NH   12
#define DH   64
#define MTOK (BQ * TT)      /* 8192 */
#define C3   (3 * CC)       /* 2304 */
#define C4   (4 * CC)       /* 3072 */

typedef __bf16 bf16;
typedef bf16  v16bf  __attribute__((ext_vector_type(16)));
typedef bf16  bf16x8 __attribute__((ext_vector_type(8)));
typedef float v8f    __attribute__((ext_vector_type(8)));

// ---------------- small helpers ----------------
__device__ __forceinline__ v8f vzero8() {
  v8f z;
#pragma unroll
  for (int i = 0; i < 8; ++i) z[i] = 0.0f;
  return z;
}

__device__ __forceinline__ v16bf pack16(bf16x8 lo, bf16x8 hi) {
  v16bf r;
#pragma unroll
  for (int i = 0; i < 8; ++i) { r[i] = lo[i]; r[i + 8] = hi[i]; }
  return r;
}

// A fragment 16(M)x32(K), row-major source with leading dim ld (elements).
// ISA 7.12.2: lanes 0-15 -> M=lane, K={kb..kb+7, kb+16..kb+23}, kb=(lane>>4)*8
__device__ __forceinline__ v16bf load_a16(const bf16* p, int ld) {
  const int lane = threadIdx.x & 31;
  const bf16* q = p + (size_t)(lane & 15) * ld + ((lane >> 4) << 3);
  bf16x8 lo = *(const bf16x8*)(q);
  bf16x8 hi = *(const bf16x8*)(q + 16);
  return pack16(lo, hi);
}

// B fragment 32(K)x16(N), row-major [K,N] source with leading dim ld.
// lane -> N column (lane&15), elements -> K={kb..kb+7, kb+16..kb+23}, kb=(lane>>4)*8.
__device__ __forceinline__ v16bf load_b16(const bf16* p, int ld) {
  const int lane = threadIdx.x & 31;
  const bf16* q = p + (lane & 15) + (size_t)((lane >> 4) << 3) * ld;
  v16bf r;
#pragma unroll
  for (int i = 0; i < 8; ++i) {
    r[i] = q[(size_t)i * ld];
    r[i + 8] = q[(size_t)(i + 16) * ld];
  }
  return r;
}

__device__ __forceinline__ v8f wmma_bf16(v16bf a, v16bf b, v8f c) {
  return __builtin_amdgcn_wmma_f32_16x16x32_bf16(false, a, false, b, (short)0, c,
                                                 false, false);
}

// CDNA5 async global->LDS copy (16B per lane), tracked by ASYNCcnt.
// VDST = LDS byte offset (low 32 bits of generic shared pointer), VADDR = 64-bit
// global address. See CDNA5 ISA.
__device__ __forceinline__ void async_g2l_b128(void* lds_ptr, const void* gptr) {
  uint32_t lds_off = (uint32_t)(uintptr_t)lds_ptr;
  asm volatile("global_load_async_to_lds_b128 %0, %1, off"
               :
               : "v"(lds_off), "v"(gptr)
               : "memory");
}

__device__ __forceinline__ void wait_asynccnt0() {
  asm volatile("s_wait_asynccnt 0x0" ::: "memory");
}

// ---------------- fp32 -> bf16 convert ----------------
__global__ void f32_to_bf16_kernel(const float* __restrict__ in,
                                   bf16* __restrict__ out, long n) {
  long i = (long)blockIdx.x * blockDim.x + threadIdx.x;
  const long stride = (long)gridDim.x * blockDim.x;
  for (; i < n; i += stride) out[i] = (bf16)in[i];
}

// ---------------- LayerNorm (one row of C=768 per block, 256 thr) -------------
__global__ __launch_bounds__(256) void layernorm_kernel(
    const float* __restrict__ x, const float* __restrict__ w,
    const float* __restrict__ b, bf16* __restrict__ out) {
  __shared__ float s1[256], s2[256];
  const int row = blockIdx.x, tid = threadIdx.x;
  const float* xr = x + (size_t)row * CC;
  float v[3], a = 0.f, q = 0.f;
#pragma unroll
  for (int i = 0; i < 3; ++i) {
    v[i] = xr[tid + i * 256];
    a += v[i];
    q += v[i] * v[i];
  }
  s1[tid] = a; s2[tid] = q;
  __syncthreads();
  for (int off = 128; off > 0; off >>= 1) {
    if (tid < off) { s1[tid] += s1[tid + off]; s2[tid] += s2[tid + off]; }
    __syncthreads();
  }
  const float mu = s1[0] * (1.0f / CC);
  const float var = s2[0] * (1.0f / CC) - mu * mu;
  const float rs = rsqrtf(var + 1e-5f);
#pragma unroll
  for (int i = 0; i < 3; ++i) {
    const int c = tid + i * 256;
    out[(size_t)row * CC + c] = (bf16)((v[i] - mu) * rs * w[c] + b[c]);
  }
}

// ---------------- WMMA GEMM: D = A[MxK] * B[KxN] + bias (+res/gelu) ----------
// BM=128 BN=128 BK=32; 256 threads = 8 waves (2x4); wave tile 64x32 = 8 WMMA.
// Double-buffered LDS filled with GLOBAL_LOAD_ASYNC_TO_LDS_B128 (ASYNCcnt).
template <bool RES, bool GELU, bool OUTB>
__global__ __launch_bounds__(256) void gemm_bf16_kernel(
    const bf16* __restrict__ A, const bf16* __restrict__ Bw,
    const float* __restrict__ bias, const float* __restrict__ resid,
    float* __restrict__ outF, bf16* __restrict__ outB, int M, int N, int K) {
  __shared__ bf16 As[2][128][48];   // 32 cols + pad; row stride 96B (16B ok)
  __shared__ bf16 Bs[2][32][136];   // 128 cols + pad; row stride 272B (16B ok)

  const int tid = threadIdx.x;
  const int lane = tid & 31, wave = tid >> 5;
  const int wm = wave >> 2, wn = wave & 3;   // 2 x 4 wave grid
  const int m0 = blockIdx.y * 128;
  const int n0 = blockIdx.x * 128;

  v8f acc[4][2];
#pragma unroll
  for (int i = 0; i < 4; ++i)
#pragma unroll
    for (int j = 0; j < 2; ++j) acc[i][j] = vzero8();

  // per-thread async-copy chunks: A 128x32 = 512 x b128, B 32x128 = 512 x b128
  const int ach0 = tid, ach1 = tid + 256;

  auto issue_tile = [&](int buf, int k0) {
#pragma unroll
    for (int j = 0; j < 2; ++j) {
      const int chunk = (j == 0) ? ach0 : ach1;
      const int arow = chunk >> 2, acol = (chunk & 3) * 8;     // A: 4 chunks/row
      async_g2l_b128(&As[buf][arow][acol],
                     &A[(size_t)(m0 + arow) * K + k0 + acol]);
      const int brow = chunk >> 4, bcol = (chunk & 15) * 8;    // B: 16 chunks/row
      async_g2l_b128(&Bs[buf][brow][bcol],
                     &Bw[(size_t)(k0 + brow) * N + n0 + bcol]);
    }
  };

  const int nk = K >> 5;
  issue_tile(0, 0);
  wait_asynccnt0();
  __syncthreads();

  for (int kk = 0; kk < nk; ++kk) {
    const int cur = kk & 1;
    if (kk + 1 < nk) issue_tile(1 - cur, (kk + 1) << 5);

    v16bf af[4];
#pragma unroll
    for (int i = 0; i < 4; ++i)
      af[i] = load_a16(&As[cur][wm * 64 + i * 16][0], 48);
    v16bf bf0 = load_b16(&Bs[cur][0][wn * 32], 136);
    v16bf bf1 = load_b16(&Bs[cur][0][wn * 32 + 16], 136);

#pragma unroll
    for (int i = 0; i < 4; ++i) {
      acc[i][0] = wmma_bf16(af[i], bf0, acc[i][0]);
      acc[i][1] = wmma_bf16(af[i], bf1, acc[i][1]);
    }

    wait_asynccnt0();   // next tile landed in LDS
    __syncthreads();    // everyone done reading cur; writes visible
  }

#pragma unroll
  for (int i = 0; i < 4; ++i)
#pragma unroll
    for (int j = 0; j < 2; ++j) {
      const int col = n0 + wn * 32 + j * 16 + (lane & 15);
      const float bv = bias[col];
#pragma unroll
      for (int r = 0; r < 8; ++r) {
        const int row = m0 + wm * 64 + i * 16 + ((lane >> 4) << 3) + r;
        float v = acc[i][j][r] + bv;
        if (RES) v += resid[(size_t)row * N + col];
        if (GELU) v = 0.5f * v * (1.0f + erff(v * 0.70710678118654752f));
        if (OUTB) outB[(size_t)row * N + col] = (bf16)v;
        else      outF[(size_t)row * N + col] = v;
      }
    }
}

// ---------------- split qkv (ref order k,q,v) into bf16 attention layouts ----
// q -> [B,H,T,D]; k -> kT [B,H,D,T]; v -> [B,H,T,D]
__global__ void split_qkv_kernel(const float* __restrict__ qkv,
                                 bf16* __restrict__ qb, bf16* __restrict__ ktb,
                                 bf16* __restrict__ vb) {
  const long n = (long)MTOK * C3;
  long i = (long)blockIdx.x * blockDim.x + threadIdx.x;
  const long stride = (long)gridDim.x * blockDim.x;
  for (; i < n; i += stride) {
    const int c3 = (int)(i % C3);
    const long bt = i / C3;
    const int t = (int)(bt % TT);
    const int b = (int)(bt / TT);
    const int chunk = c3 / CC;
    const int c = c3 % CC;
    const int h = c / DH, d = c % DH;
    const bf16 v = (bf16)qkv[i];
    if (chunk == 0)
      ktb[(((size_t)b * NH + h) * DH + d) * TT + t] = v;          // k (transposed)
    else if (chunk == 1)
      qb[(((size_t)b * NH + h) * TT + t) * DH + d] = v;           // q
    else
      vb[(((size_t)b * NH + h) * TT + t) * DH + d] = v;           // v
  }
}

// ---------------- causal flash attention, no 1/sqrt(d) scale -----------------
// grid: (B*NH, TT/64); block: 128 (4 waves). Wave w owns q rows [qbase,qbase+16).
__global__ __launch_bounds__(128) void attention_kernel(
    const bf16* __restrict__ qb, const bf16* __restrict__ ktb,
    const bf16* __restrict__ vb, bf16* __restrict__ yb) {
  __shared__ bf16 Ps[4][16][72];  // per-wave P staging; row stride 144B (16B ok)

  const int lane = threadIdx.x & 31, w = threadIdx.x >> 5;
  const int bh = blockIdx.x, qblk = blockIdx.y;
  const int b = bh / NH, h = bh % NH;
  const int qbase = qblk * 64 + w * 16;
  const int rowoff = (lane >> 4) << 3;  // 0 or 8
  const int nc = lane & 15;

  const bf16* qptr = qb + ((size_t)bh * TT + qbase) * DH;
  const v16bf aq0 = load_a16(qptr, DH);       // d = 0..31
  const v16bf aq1 = load_a16(qptr + 32, DH);  // d = 32..63

  float m[8], l[8];
  v8f o[4];
#pragma unroll
  for (int r = 0; r < 8; ++r) { m[r] = -INFINITY; l[r] = 0.0f; }
#pragma unroll
  for (int n = 0; n < 4; ++n) o[n] = vzero8();

  for (int kb = 0; kb <= qblk; ++kb) {
    const int key0 = kb * 64;

    // S = Q * K^T  (per wave: 16 q rows x 64 keys)
    v8f s[4];
#pragma unroll
    for (int n = 0; n < 4; ++n) {
      const bf16* kp = ktb + (size_t)bh * DH * TT + key0 + n * 16;
      v16bf bk0 = load_b16(kp, TT);                    // d rows 0..31
      v16bf bk1 = load_b16(kp + (size_t)32 * TT, TT);  // d rows 32..63
      s[n] = wmma_bf16(aq0, bk0, vzero8());
      s[n] = wmma_bf16(aq1, bk1, s[n]);
    }

    // causal mask + online softmax (row reductions over 16-lane groups)
#pragma unroll
    for (int r = 0; r < 8; ++r) {
      const int trow = qbase + rowoff + r;
      float mx = -INFINITY;
#pragma unroll
      for (int n = 0; n < 4; ++n) {
        const int col = key0 + n * 16 + nc;
        const float sv = (col <= trow) ? s[n][r] : -INFINITY;
        s[n][r] = sv;
        mx = fmaxf(mx, sv);
      }
#pragma unroll
      for (int off = 1; off < 16; off <<= 1)
        mx = fmaxf(mx, __shfl_xor(mx, off, 32));
      const float mnew = fmaxf(m[r], mx);
      const float scale = __expf(m[r] - mnew);
      float rsum = 0.0f;
#pragma unroll
      for (int n = 0; n < 4; ++n) {
        const float p = __expf(s[n][r] - mnew);
        rsum += p;
        Ps[w][rowoff + r][n * 16 + nc] = (bf16)p;
      }
#pragma unroll
      for (int off = 1; off < 16; off <<= 1) rsum += __shfl_xor(rsum, off, 32);
      l[r] = l[r] * scale + rsum;
      m[r] = mnew;
#pragma unroll
      for (int n = 0; n < 4; ++n) o[n][r] *= scale;
    }
    __syncthreads();

    // O += P * V   (P re-fragmented through LDS as A operand)
    const v16bf ap0 = load_a16(&Ps[w][0][0], 72);   // keys 0..31 of block
    const v16bf ap1 = load_a16(&Ps[w][0][32], 72);  // keys 32..63 of block
#pragma unroll
    for (int n = 0; n < 4; ++n) {
      const bf16* vp = vb + ((size_t)bh * TT + key0) * DH + n * 16;
      v16bf bv0 = load_b16(vp, DH);                    // key rows 0..31
      v16bf bv1 = load_b16(vp + (size_t)32 * DH, DH);  // key rows 32..63
      o[n] = wmma_bf16(ap0, bv0, o[n]);
      o[n] = wmma_bf16(ap1, bv1, o[n]);
    }
    __syncthreads();
  }

  // normalize and store y in [B, T, H*D] bf16 (feeds out-proj GEMM directly)
#pragma unroll
  for (int r = 0; r < 8; ++r) {
    const int trow = qbase + rowoff + r;
    const float inv = 1.0f / l[r];
#pragma unroll
    for (int n = 0; n < 4; ++n)
      yb[(((size_t)b * TT + trow) * NH + h) * DH + n * 16 + nc] =
          (bf16)(o[n][r] * inv);
  }
}

// ---------------- launch orchestration ----------------
extern "C" void kernel_launch(void* const* d_in, const int* in_sizes, int n_in,
                              void* d_out, int out_size, void* d_ws,
                              size_t ws_size, hipStream_t stream) {
  const float* x     = (const float*)d_in[0];
  const float* ln_w  = (const float*)d_in[1];
  const float* ln_b  = (const float*)d_in[2];
  const float* w_qkv = (const float*)d_in[3];
  const float* b_qkv = (const float*)d_in[4];
  const float* w_out = (const float*)d_in[5];
  const float* b_out = (const float*)d_in[6];
  const float* w_c1  = (const float*)d_in[7];
  const float* b_c1  = (const float*)d_in[8];
  const float* w_c2  = (const float*)d_in[9];
  const float* b_c2  = (const float*)d_in[10];
  float* out = (float*)d_out;

  char* ws = (char*)d_ws;
  auto alloc = [&](size_t bytes) -> char* {
    char* p = ws;
    ws += (bytes + 255) & ~(size_t)255;
    return p;
  };

  bf16* wqkvb = (bf16*)alloc((size_t)CC * C3 * 2);
  bf16* woutb = (bf16*)alloc((size_t)CC * CC * 2);
  bf16* wc1b  = (bf16*)alloc((size_t)CC * C4 * 2);
  bf16* wc2b  = (bf16*)alloc((size_t)C4 * CC * 2);
  bf16* hbf   = (bf16*)alloc((size_t)MTOK * CC * 2);  // LN output (reused)
  char* big   = alloc((size_t)MTOK * C3 * 4);         // qkv f32, later mlp bf16
  float* qkvF = (float*)big;
  bf16*  mbf  = (bf16*)big;
  bf16* qb  = (bf16*)alloc((size_t)MTOK * CC * 2);
  bf16* ktb = (bf16*)alloc((size_t)MTOK * CC * 2);
  bf16* vb  = (bf16*)alloc((size_t)MTOK * CC * 2);
  bf16* ybf = (bf16*)alloc((size_t)MTOK * CC * 2);
  float* x2 = (float*)alloc((size_t)MTOK * CC * 4);

  // 1) weights -> bf16
  f32_to_bf16_kernel<<<1024, 256, 0, stream>>>(w_qkv, wqkvb, (long)CC * C3);
  f32_to_bf16_kernel<<<1024, 256, 0, stream>>>(w_out, woutb, (long)CC * CC);
  f32_to_bf16_kernel<<<1024, 256, 0, stream>>>(w_c1, wc1b, (long)CC * C4);
  f32_to_bf16_kernel<<<1024, 256, 0, stream>>>(w_c2, wc2b, (long)C4 * CC);

  // 2) h = LN(x)
  layernorm_kernel<<<MTOK, 256, 0, stream>>>(x, ln_w, ln_b, hbf);

  // 3) qkv = h @ w_qkv + b_qkv
  gemm_bf16_kernel<false, false, false>
      <<<dim3(C3 / 128, MTOK / 128), 256, 0, stream>>>(
          hbf, wqkvb, b_qkv, nullptr, qkvF, nullptr, MTOK, C3, CC);

  // 4) split into k (transposed), q, v bf16
  split_qkv_kernel<<<4096, 256, 0, stream>>>(qkvF, qb, ktb, vb);

  // 5) y = causal_attention(q, k, v)
  attention_kernel<<<dim3(BQ * NH, TT / 64), 128, 0, stream>>>(qb, ktb, vb, ybf);

  // 6) x2 = x + y @ w_out + b_out
  gemm_bf16_kernel<true, false, false>
      <<<dim3(CC / 128, MTOK / 128), 256, 0, stream>>>(
          ybf, woutb, b_out, x, x2, nullptr, MTOK, CC, CC);

  // 7) h2 = LN(x2)   (same ln params per reference)
  layernorm_kernel<<<MTOK, 256, 0, stream>>>(x2, ln_w, ln_b, hbf);

  // 8) m = gelu(h2 @ w_c1 + b_c1)  -> bf16
  gemm_bf16_kernel<false, true, true>
      <<<dim3(C4 / 128, MTOK / 128), 256, 0, stream>>>(
          hbf, wc1b, b_c1, nullptr, nullptr, mbf, MTOK, C4, CC);

  // 9) out = x2 + m @ w_c2 + b_c2
  gemm_bf16_kernel<true, false, false>
      <<<dim3(CC / 128, MTOK / 128), 256, 0, stream>>>(
          mbf, wc2b, b_c2, x2, out, nullptr, MTOK, CC, C4);
}
